// AttentionHead_23227183137189
// MI455X (gfx1250) — matrix operand order repaired and checked by reading, not compiled, not run on hardware
//
#include <hip/hip_runtime.h>

// ---------------------------------------------------------------------------
// Single-head causal attention for MI455X (gfx1250, wave32, WMMA bf16).
// B=8, T=2048, C=1024, H=128.  Three phases:
//   1) weight transpose+convert fp32 -> bf16  (Wt[h][c])
//   2) QKV projection GEMM via v_wmma_f32_16x16x32_bf16 (V stored transposed)
//   3) flash attention, one wave per 16-query tile, online softmax,
//      P re-layout via per-wave LDS slice (no inter-wave sync needed).
// ---------------------------------------------------------------------------

constexpr int B_ = 8;
constexpr int T_ = 2048;
constexpr int C_ = 1024;
constexpr int H_ = 128;

typedef __attribute__((ext_vector_type(16))) __bf16       v16bf;
typedef __attribute__((ext_vector_type(2)))  __bf16       v2bf;
typedef __attribute__((ext_vector_type(8)))  float        v8f;
typedef __attribute__((ext_vector_type(4)))  unsigned int u32x4;
typedef __attribute__((ext_vector_type(4)))  float        f32x4;

union Frag16 {
  v16bf          bf;
  u32x4          q[2];
  unsigned int   u[8];
  unsigned short h[16];
};

#if defined(__has_builtin)
#if __has_builtin(__builtin_amdgcn_cvt_pk_bf16_f32)
#define HAVE_CVT_PK_BF16 1
#endif
#endif

static __device__ __forceinline__ unsigned short f32_to_bf16(float f) {
  unsigned int u = __float_as_uint(f);
  unsigned int r = u + 0x7FFFu + ((u >> 16) & 1u);   // round-to-nearest-even
  return (unsigned short)(r >> 16);
}

static __device__ __forceinline__ unsigned int pack2_bf16(float lo, float hi) {
#ifdef HAVE_CVT_PK_BF16
  union { v2bf v; unsigned int u; } r;
  r.v = __builtin_amdgcn_cvt_pk_bf16_f32(lo, hi);    // v_cvt_pk_bf16_f32
  return r.u;
#else
  unsigned int ul = __float_as_uint(lo);
  ul = (ul + 0x7FFFu + ((ul >> 16) & 1u)) >> 16;
  unsigned int uh = __float_as_uint(hi);
  uh = (uh + 0x7FFFu + ((uh >> 16) & 1u)) & 0xFFFF0000u;
  return ul | uh;
#endif
}

static __device__ __forceinline__ v8f wmma_bf16(const Frag16& a, const Frag16& b, v8f c) {
  // D = A(16x32 bf16) * B(32x16 bf16) + C(16x16 f32)
  return __builtin_amdgcn_wmma_f32_16x16x32_bf16(
      /*neg_a=*/false, a.bf, /*neg_b=*/false, b.bf,
      /*c_mod=*/(short)0, c, /*reuse_a=*/false, /*reuse_b=*/false);
}

// ---------------------------------------------------------------------------
// Kernel 1: W[c][h] fp32  ->  Wt[w][h][c] bf16  (3 weight matrices)
// ---------------------------------------------------------------------------
__global__ __launch_bounds__(256) void wt_convert_kernel(
    const float* __restrict__ Wq, const float* __restrict__ Wk,
    const float* __restrict__ Wv, unsigned short* __restrict__ Wt) {
  int idx = blockIdx.x * 256 + threadIdx.x;
  if (idx >= 3 * C_ * H_) return;
  int w = idx / (C_ * H_);
  int r = idx - w * (C_ * H_);
  int c = r / H_;
  int h = r - c * H_;
  const float* W = (w == 0) ? Wq : (w == 1) ? Wk : Wv;
  Wt[w * (H_ * C_) + h * C_ + c] = f32_to_bf16(W[c * H_ + h]);
}

// ---------------------------------------------------------------------------
// Kernel 2: QKV projection.  One wave per 16 rows of x (BT=16384 rows).
// Q,K row-major [BT][H] bf16; V stored transposed per batch: Vt[b][H][T] bf16.
// ---------------------------------------------------------------------------
__global__ __launch_bounds__(128) void qkv_proj_kernel(
    const float* __restrict__ x, const unsigned short* __restrict__ Wt,
    unsigned short* __restrict__ Q, unsigned short* __restrict__ K,
    unsigned short* __restrict__ Vt) {
  const int lane = threadIdx.x & 31;
  const int wave = blockIdx.x * 4 + (threadIdx.x >> 5);
  const int n    = lane & 15;      // A-row / B-col within tile
  const int hfl  = lane >> 4;      // lane half
  const int rowbase = wave * 16;   // row block in [0, B*T)

  const float* xrow = x + (size_t)(rowbase + n) * C_;

  for (int w = 0; w < 3; ++w) {
    v8f acc[8];
#pragma unroll
    for (int t = 0; t < 8; ++t)
      acc[t] = (v8f){0.f, 0.f, 0.f, 0.f, 0.f, 0.f, 0.f, 0.f};
    const unsigned short* Wb = Wt + w * (H_ * C_);

    for (int ks = 0; ks < C_ / 32; ++ks) {
      const int cb = ks * 32;
      // A fragment from x (fp32 -> bf16 in registers).
      // Lane (m=n, half=hfl) holds K run [cb+16h, +16).
      f32x4 a0 = *(const f32x4*)(xrow + cb + 16 * hfl);
      f32x4 a1 = *(const f32x4*)(xrow + cb + 16 * hfl + 4);
      f32x4 a2 = *(const f32x4*)(xrow + cb + 16 * hfl + 8);
      f32x4 a3 = *(const f32x4*)(xrow + cb + 16 * hfl + 12);
      Frag16 a;
      a.u[0] = pack2_bf16(a0.x, a0.y);  a.u[1] = pack2_bf16(a0.z, a0.w);
      a.u[2] = pack2_bf16(a1.x, a1.y);  a.u[3] = pack2_bf16(a1.z, a1.w);
      a.u[4] = pack2_bf16(a2.x, a2.y);  a.u[5] = pack2_bf16(a2.z, a2.w);
      a.u[6] = pack2_bf16(a3.x, a3.y);  a.u[7] = pack2_bf16(a3.z, a3.w);

#pragma unroll
      for (int nt = 0; nt < 8; ++nt) {
        // B fragment: lane holds 16 contiguous K at 16h of row (16nt+n) of Wt.
        const unsigned short* br = Wb + (size_t)(nt * 16 + n) * C_ + cb + 16 * hfl;
        Frag16 bf;
        bf.q[0] = *(const u32x4*)(br);
        bf.q[1] = *(const u32x4*)(br + 8);
        acc[nt] = wmma_bf16(a, bf, acc[nt]);
      }
    }

    if (w < 2) {
      unsigned short* dst = (w == 0) ? Q : K;
#pragma unroll
      for (int nt = 0; nt < 8; ++nt)
#pragma unroll
        for (int j = 0; j < 8; ++j)
          dst[(size_t)(rowbase + j + 8 * hfl) * H_ + nt * 16 + n] =
              f32_to_bf16(acc[nt][j]);
    } else {
      const int bidx  = rowbase >> 11;            // / T_
      const int tbase = rowbase - (bidx << 11);
      unsigned short* dst = Vt + (size_t)bidx * (H_ * T_);
#pragma unroll
      for (int nt = 0; nt < 8; ++nt)
#pragma unroll
        for (int j = 0; j < 8; ++j)
          dst[(size_t)(nt * 16 + n) * T_ + tbase + j + 8 * hfl] =
              f32_to_bf16(acc[nt][j]);
    }
  }
}

// ---------------------------------------------------------------------------
// Kernel 3: flash attention.  One wave per 16-query tile (1024 waves).
// ---------------------------------------------------------------------------
__global__ __launch_bounds__(128) void flash_attn_kernel(
    const unsigned short* __restrict__ Q, const unsigned short* __restrict__ K,
    const unsigned short* __restrict__ Vt, float* __restrict__ out) {
  __shared__ __align__(16) unsigned short sP[4 * 16 * 32];  // per-wave 1KB slice

  const int lane = threadIdx.x & 31;
  const int wv   = threadIdx.x >> 5;
  const int wave = blockIdx.x * 4 + wv;
  const int n    = lane & 15;
  const int hfl  = lane >> 4;
  const int b    = wave >> 7;         // / (T/16)
  const int qbase = (wave & 127) * 16;
  unsigned short* sPl = sP + wv * (16 * 32);

  // Resident Q fragments (A layout), K-dim = H = 128 -> 4 fragments.
  const unsigned short* qrow = Q + (size_t)(b * T_ + qbase + n) * H_;
  Frag16 qf[4];
#pragma unroll
  for (int ks = 0; ks < 4; ++ks) {
    qf[ks].q[0] = *(const u32x4*)(qrow + ks * 32 + 16 * hfl);
    qf[ks].q[1] = *(const u32x4*)(qrow + ks * 32 + 16 * hfl + 8);
  }
  const unsigned short* Kb = K + (size_t)b * T_ * H_;
  const unsigned short* Vb = Vt + (size_t)b * H_ * T_;

  v8f o[8];
#pragma unroll
  for (int t = 0; t < 8; ++t)
    o[t] = (v8f){0.f, 0.f, 0.f, 0.f, 0.f, 0.f, 0.f, 0.f};
  float rmax[8], rsum[8];
  int   mrow[8];
#pragma unroll
  for (int j = 0; j < 8; ++j) {
    rmax[j] = -1e30f;
    rsum[j] = 0.f;
    mrow[j] = qbase + j + 8 * hfl;   // global query row for C/D vgpr j
  }

  const float scale = 0.08838834764831845f;     // 1/sqrt(128)
  const int ntiles  = (qbase + 16 + 31) >> 5;   // causal: keys [0, qbase+16)

  for (int t = 0; t < ntiles; ++t) {
    const int kvb = t * 32;
    v8f s0 = (v8f){0.f, 0.f, 0.f, 0.f, 0.f, 0.f, 0.f, 0.f};
    v8f s1 = (v8f){0.f, 0.f, 0.f, 0.f, 0.f, 0.f, 0.f, 0.f};
    // S = Q * K^T   (B fragment: lane = key (kvb + 16*ntile + n), contiguous h)
#pragma unroll
    for (int ks = 0; ks < 4; ++ks) {
      const unsigned short* kr0 = Kb + (size_t)(kvb + n) * H_      + ks * 32 + 16 * hfl;
      const unsigned short* kr1 = Kb + (size_t)(kvb + 16 + n) * H_ + ks * 32 + 16 * hfl;
      Frag16 b0, b1;
      b0.q[0] = *(const u32x4*)(kr0);  b0.q[1] = *(const u32x4*)(kr0 + 8);
      b1.q[0] = *(const u32x4*)(kr1);  b1.q[1] = *(const u32x4*)(kr1 + 8);
      s0 = wmma_bf16(qf[ks], b0, s0);
      s1 = wmma_bf16(qf[ks], b1, s1);
    }

    // Online softmax (per C/D row j + 8*hfl; reduce across 16-lane group).
    float alpha[8];
#pragma unroll
    for (int j = 0; j < 8; ++j) {
      float v0 = s0[j] * scale;  if (kvb + n      > mrow[j]) v0 = -1e30f;
      float v1 = s1[j] * scale;  if (kvb + 16 + n > mrow[j]) v1 = -1e30f;
      float tm = fmaxf(v0, v1);
      tm = fmaxf(tm, __shfl_xor(tm, 1, 32));
      tm = fmaxf(tm, __shfl_xor(tm, 2, 32));
      tm = fmaxf(tm, __shfl_xor(tm, 4, 32));
      tm = fmaxf(tm, __shfl_xor(tm, 8, 32));
      float nm = fmaxf(rmax[j], tm);
      alpha[j] = __expf(rmax[j] - nm);
      rmax[j]  = nm;
      float p0 = __expf(v0 - nm);
      float p1 = __expf(v1 - nm);
      // Stage P row-major 16x32 bf16 into this wave's LDS slice.
      sPl[(j + 8 * hfl) * 32 + n]      = f32_to_bf16(p0);
      sPl[(j + 8 * hfl) * 32 + 16 + n] = f32_to_bf16(p1);
      float ts = p0 + p1;
      ts += __shfl_xor(ts, 1, 32);
      ts += __shfl_xor(ts, 2, 32);
      ts += __shfl_xor(ts, 4, 32);
      ts += __shfl_xor(ts, 8, 32);
      rsum[j] = rsum[j] * alpha[j] + ts;
    }
#pragma unroll
    for (int nt = 0; nt < 8; ++nt)
#pragma unroll
      for (int j = 0; j < 8; ++j)
        o[nt][j] *= alpha[j];

    // LDS ops are in-order per wave; wait for the b16 stores, then read the
    // P tile back in WMMA A layout (two b128 per lane).
    asm volatile("s_wait_dscnt 0" ::: "memory");
    Frag16 pa;
    pa.q[0] = *(const u32x4*)(sPl + n * 32 + 16 * hfl);
    pa.q[1] = *(const u32x4*)(sPl + n * 32 + 16 * hfl + 8);

    // O += P * V   (B fragment from transposed V: lane = head col, contiguous s)
#pragma unroll
    for (int nt = 0; nt < 8; ++nt) {
      const unsigned short* vr = Vb + (size_t)(nt * 16 + n) * T_ + kvb + 16 * hfl;
      Frag16 bv;
      bv.q[0] = *(const u32x4*)(vr);  bv.q[1] = *(const u32x4*)(vr + 8);
      o[nt] = wmma_bf16(pa, bv, o[nt]);
    }
  }

  float inv[8];
#pragma unroll
  for (int j = 0; j < 8; ++j) inv[j] = 1.0f / rsum[j];
  float* orow = out + (size_t)(b * T_ + qbase) * H_;
#pragma unroll
  for (int nt = 0; nt < 8; ++nt)
#pragma unroll
    for (int j = 0; j < 8; ++j)
      __builtin_nontemporal_store(o[nt][j] * inv[j],
                                  orow + (size_t)(j + 8 * hfl) * H_ + nt * 16 + n);
}

// ---------------------------------------------------------------------------
// Host side
// ---------------------------------------------------------------------------
extern "C" void kernel_launch(void* const* d_in, const int* in_sizes, int n_in,
                              void* d_out, int out_size, void* d_ws, size_t ws_size,
                              hipStream_t stream) {
  const float* x  = (const float*)d_in[0];
  const float* Wq = (const float*)d_in[1];
  const float* Wk = (const float*)d_in[2];
  const float* Wv = (const float*)d_in[3];
  float* out = (float*)d_out;

  // Workspace layout (bf16): Wt[3][H][C] | Q[BT][H] | K[BT][H] | Vt[B][H][T]
  char* ws = (char*)d_ws;
  const size_t wtBytes  = (size_t)3 * H_ * C_ * 2;           // 768 KB
  const size_t matBytes = (size_t)B_ * T_ * H_ * 2;          // 4 MB each
  unsigned short* Wt = (unsigned short*)ws;
  unsigned short* Qb = (unsigned short*)(ws + wtBytes);
  unsigned short* Kb = (unsigned short*)(ws + wtBytes + matBytes);
  unsigned short* Vt = (unsigned short*)(ws + wtBytes + 2 * matBytes);

  wt_convert_kernel<<<(3 * C_ * H_ + 255) / 256, 256, 0, stream>>>(Wq, Wk, Wv, Wt);

  const int nWaves = (B_ * T_) / 16;                          // 1024
  qkv_proj_kernel<<<nWaves / 4, 128, 0, stream>>>(x, Wt, Qb, Kb, Vt);

  flash_attn_kernel<<<nWaves / 4, 128, 0, stream>>>(Qb, Kb, Vt, out);
}
